// DiTSelfAttention_54443005444352
// MI455X (gfx1250) — matrix operand order repaired and checked by reading, not compiled, not run on hardware
//
#include <hip/hip_runtime.h>
#include <stdint.h>

// ---------------------------------------------------------------------------
// DiT self-attention for MI455X (gfx1250, wave32, WMMA).
// B=2, S=2048, H=1024, NH=16, HD=64.  bf16 compute, f32 accumulate.
// All GEMMs + flash attention use v_wmma_f32_16x16x32_bf16 with
// double-buffered LDS staging (one barrier per K-step).
// ---------------------------------------------------------------------------

typedef __bf16 bf16_t;
typedef __bf16 v16bf __attribute__((ext_vector_type(16)));
typedef float  v8f   __attribute__((ext_vector_type(8)));

union FragAB { v16bf v; uint4 q[2]; };

__device__ __forceinline__ v8f wmma_bf16(const FragAB& a, const FragAB& b, v8f c) {
    return __builtin_amdgcn_wmma_f32_16x16x32_bf16(false, a.v, false, b.v, (short)0, c, false, false);
}

// ---------------------------------------------------------------------------
// f32 -> bf16 conversion (vectorized 4 elems / thread)
// ---------------------------------------------------------------------------
__global__ void dit_cvt_bf16(const float* __restrict__ src, bf16_t* __restrict__ dst, int n4) {
    int i = blockIdx.x * blockDim.x + threadIdx.x;
    if (i < n4) {
        float4 f = ((const float4*)src)[i];
        union { bf16_t h[4]; uint2 u; } pk;
        pk.h[0] = (bf16_t)f.x; pk.h[1] = (bf16_t)f.y;
        pk.h[2] = (bf16_t)f.z; pk.h[3] = (bf16_t)f.w;
        ((uint2*)dst)[i] = pk.u;
    }
}

// ---------------------------------------------------------------------------
// Shared GEMM geometry: block 128(M) x 128(N), K-step 32, 8 waves (2x4),
// wave tile 64x32 -> acc[4][2], 12 ds_load_b128 feed 8 WMMA per K-step.
// LDS double buffered: one barrier per K-step.
// ---------------------------------------------------------------------------

// QKV projection: X[4096,1024](bf16) @ W^T + b -> Q/K/V [b][h][s][d] (bf16).
// Q gets (1/8)*log2(e) folded in (softmax later runs in exp2 domain).
__global__ __launch_bounds__(256) void dit_gemm_qkv(
    const bf16_t* __restrict__ X,
    const bf16_t* __restrict__ Wq, const bf16_t* __restrict__ Wk, const bf16_t* __restrict__ Wv,
    const float* __restrict__ bq, const float* __restrict__ bk, const float* __restrict__ bv,
    bf16_t* __restrict__ Q, bf16_t* __restrict__ K, bf16_t* __restrict__ V)
{
    __shared__ __align__(16) bf16_t sX[2][128][32];  // 16 KB
    __shared__ __align__(16) bf16_t sW[2][128][32];  // 16 KB

    const int m0    = blockIdx.x * 128;
    const int nTile = blockIdx.y;            // 0..23 : 8 tiles each for Q,K,V
    const int which = nTile >> 3;
    const int n0    = (nTile & 7) * 128;

    const bf16_t* W    = (which == 0) ? Wq : (which == 1) ? Wk : Wv;
    const float*  bias = (which == 0) ? bq : (which == 1) ? bk : bv;
    bf16_t*       Out  = (which == 0) ? Q  : (which == 1) ? K  : V;
    const float scale  = (which == 0) ? 0.18033688f : 1.0f;   // 0.125 * log2(e)

    const int t    = threadIdx.x;
    const int lane = t & 31;
    const int w    = t >> 5;       // 0..7
    const int wm   = w >> 2;       // 0..1  (64 rows each)
    const int wn   = w & 3;        // 0..3  (32 cols each)
    const int l16  = lane & 15;
    const int half = lane >> 4;

    const int srow = t >> 1;            // 0..127
    const int sseg = (t & 1) * 16;      // 0 or 16
    const bf16_t* gX = X + (size_t)(m0 + srow) * 1024 + sseg;
    const bf16_t* gW = W + (size_t)(n0 + srow) * 1024 + sseg;

    v8f acc[4][2];
    #pragma unroll
    for (int i = 0; i < 4; ++i)
        #pragma unroll
        for (int j = 0; j < 2; ++j)
            #pragma unroll
            for (int e = 0; e < 8; ++e) acc[i][j][e] = 0.0f;

    uint4 rx0 = *(const uint4*)(gX);
    uint4 rx1 = *(const uint4*)(gX + 8);
    uint4 rw0 = *(const uint4*)(gW);
    uint4 rw1 = *(const uint4*)(gW + 8);

    for (int ks = 0; ks < 32; ++ks) {
        const int buf = ks & 1;
        *(uint4*)&sX[buf][srow][sseg]     = rx0;
        *(uint4*)&sX[buf][srow][sseg + 8] = rx1;
        *(uint4*)&sW[buf][srow][sseg]     = rw0;
        *(uint4*)&sW[buf][srow][sseg + 8] = rw1;
        if (ks + 1 < 32) {
            const int k1 = (ks + 1) * 32;
            rx0 = *(const uint4*)(gX + k1);
            rx1 = *(const uint4*)(gX + k1 + 8);
            rw0 = *(const uint4*)(gW + k1);
            rw1 = *(const uint4*)(gW + k1 + 8);
            if (ks + 2 < 32) {                       // warm L2 for K-step+2
                __builtin_prefetch(gX + k1 + 32, 0, 0);
                __builtin_prefetch(gW + k1 + 32, 0, 0);
            }
        }
        __syncthreads();

        FragAB a[4], b[2];
        #pragma unroll
        for (int i = 0; i < 4; ++i) {            // A: lane=row, elems k = 8*half+e (+16)
            int row = wm * 64 + i * 16 + l16;
            a[i].q[0] = *(const uint4*)&sX[buf][row][half * 8];
            a[i].q[1] = *(const uint4*)&sX[buf][row][16 + half * 8];
        }
        #pragma unroll
        for (int j = 0; j < 2; ++j) {            // B: lane=col, elems k = 8*half+e (+16)
            int col = wn * 32 + j * 16 + l16;
            b[j].q[0] = *(const uint4*)&sW[buf][col][half * 8];
            b[j].q[1] = *(const uint4*)&sW[buf][col][16 + half * 8];
        }
        #pragma unroll
        for (int i = 0; i < 4; ++i)
            #pragma unroll
            for (int j = 0; j < 2; ++j)
                acc[i][j] = wmma_bf16(a[i], b[j], acc[i][j]);
    }

    // Epilogue: bias, scale, scatter to [b][h][s][d]
    #pragma unroll
    for (int i = 0; i < 4; ++i) {
        #pragma unroll
        for (int j = 0; j < 2; ++j) {
            int col = n0 + wn * 32 + j * 16 + l16;
            float bb = bias[col];
            int h = col >> 6, d = col & 63;
            #pragma unroll
            for (int r = 0; r < 8; ++r) {        // D: row M = r + 8*half
                int m  = m0 + wm * 64 + i * 16 + half * 8 + r;
                int bi = m >> 11, s = m & 2047;
                float v = (acc[i][j][r] + bb) * scale;
                Out[(((size_t)(bi * 16 + h)) * 2048 + s) * 64 + d] = (bf16_t)v;
            }
        }
    }
}

// ---------------------------------------------------------------------------
// Flash attention: one (b,h) x 64 Q-rows per block, 4 waves x 16 rows.
// K/V tiles double buffered in LDS; V stored transposed via packed b32;
// exp2-domain online softmax; P round-trips via LDS for D->A layout.
// ---------------------------------------------------------------------------
__global__ __launch_bounds__(128) void dit_attn(
    const bf16_t* __restrict__ Q, const bf16_t* __restrict__ Kb,
    const bf16_t* __restrict__ Vb, bf16_t* __restrict__ Ctx)
{
    __shared__ __align__(16) bf16_t       sK[2][64][64];    // [key][d]    16 KB
    __shared__ __align__(16) unsigned int sVt[2][64][32];   // [d][kvpair] 16 KB
    __shared__ __align__(16) bf16_t       sP[64][64];       // [qrow][key]  8 KB

    const int q0 = blockIdx.x * 64;
    const int bh = blockIdx.y;
    const int t    = threadIdx.x;
    const int lane = t & 31;
    const int w    = t >> 5;      // 0..3
    const int l16  = lane & 15;
    const int half = lane >> 4;

    const bf16_t* Qbh = Q  + (size_t)bh * 2048 * 64;
    const bf16_t* Kbh = Kb + (size_t)bh * 2048 * 64;
    const bf16_t* Vbh = Vb + (size_t)bh * 2048 * 64;

    // Q fragments (A layout), in registers for the whole kv loop.
    FragAB qa[2];
    {
        const bf16_t* qrow = Qbh + (size_t)(q0 + w * 16 + l16) * 64;
        #pragma unroll
        for (int kk = 0; kk < 2; ++kk) {
            qa[kk].q[0] = *(const uint4*)(qrow + kk * 32 + half * 8);
            qa[kk].q[1] = *(const uint4*)(qrow + kk * 32 + 16 + half * 8);
        }
    }

    // staging address maps
    const int krow = t >> 1,  kseg = (t & 1) * 32;   // K: [key][d], 64 B/thread
    const int vp   = t >> 2,  vseg = (t & 3) * 16;   // V: kv rows 2vp,2vp+1, 16 d's

    const bf16_t* gK = Kbh + krow * 64 + kseg;
    const bf16_t* gV = Vbh + (vp * 2) * 64 + vseg;

    uint4 kr[4];
    union RowU { uint4 q[2]; unsigned short s[16]; } va, vbr;
    kr[0] = *(const uint4*)(gK);
    kr[1] = *(const uint4*)(gK + 8);
    kr[2] = *(const uint4*)(gK + 16);
    kr[3] = *(const uint4*)(gK + 24);
    va.q[0]  = *(const uint4*)(gV);
    va.q[1]  = *(const uint4*)(gV + 8);
    vbr.q[0] = *(const uint4*)(gV + 64);
    vbr.q[1] = *(const uint4*)(gV + 72);

    v8f o[4];
    float mrow[8], lrow[8];
    #pragma unroll
    for (int nt = 0; nt < 4; ++nt)
        #pragma unroll
        for (int e = 0; e < 8; ++e) o[nt][e] = 0.0f;
    #pragma unroll
    for (int r = 0; r < 8; ++r) { mrow[r] = -3.0e38f; lrow[r] = 0.0f; }

    for (int kt = 0; kt < 32; ++kt) {
        const int buf = kt & 1;
        // ---- stage current tile into buf ----
        *(uint4*)&sK[buf][krow][kseg]      = kr[0];
        *(uint4*)&sK[buf][krow][kseg + 8]  = kr[1];
        *(uint4*)&sK[buf][krow][kseg + 16] = kr[2];
        *(uint4*)&sK[buf][krow][kseg + 24] = kr[3];
        #pragma unroll
        for (int e = 0; e < 16; ++e)            // packed transpose: b32 stores
            sVt[buf][vseg + e][vp] = (unsigned int)va.s[e] | ((unsigned int)vbr.s[e] << 16);
        // ---- issue next tile's global loads ----
        if (kt + 1 < 32) {
            const size_t adv = (size_t)(kt + 1) * 64 * 64;
            kr[0] = *(const uint4*)(gK + adv);
            kr[1] = *(const uint4*)(gK + adv + 8);
            kr[2] = *(const uint4*)(gK + adv + 16);
            kr[3] = *(const uint4*)(gK + adv + 24);
            va.q[0]  = *(const uint4*)(gV + adv);
            va.q[1]  = *(const uint4*)(gV + adv + 8);
            vbr.q[0] = *(const uint4*)(gV + adv + 64);
            vbr.q[1] = *(const uint4*)(gV + adv + 72);
        }
        __syncthreads();

        // ---- scores S = Q * K^T (scale already folded into Q) ----
        v8f sc[4];
        #pragma unroll
        for (int nt = 0; nt < 4; ++nt) {
            #pragma unroll
            for (int e = 0; e < 8; ++e) sc[nt][e] = 0.0f;
            #pragma unroll
            for (int kk = 0; kk < 2; ++kk) {
                FragAB kf;
                int key = nt * 16 + l16;
                kf.q[0] = *(const uint4*)&sK[buf][key][kk * 32 + half * 8];
                kf.q[1] = *(const uint4*)&sK[buf][key][kk * 32 + 16 + half * 8];
                sc[nt] = wmma_bf16(qa[kk], kf, sc[nt]);
            }
        }

        // ---- online softmax (rows M = r + 8*half across 16-lane half) ----
        float mnew[8];
        #pragma unroll
        for (int r = 0; r < 8; ++r) {
            float mx = fmaxf(fmaxf(sc[0][r], sc[1][r]), fmaxf(sc[2][r], sc[3][r]));
            mx = fmaxf(mx, __shfl_xor(mx, 1));
            mx = fmaxf(mx, __shfl_xor(mx, 2));
            mx = fmaxf(mx, __shfl_xor(mx, 4));
            mx = fmaxf(mx, __shfl_xor(mx, 8));
            mnew[r] = fmaxf(mrow[r], mx);
        }
        #pragma unroll
        for (int r = 0; r < 8; ++r) {
            float corr = exp2f(mrow[r] - mnew[r]);
            lrow[r] *= corr;
            #pragma unroll
            for (int nt = 0; nt < 4; ++nt) o[nt][r] *= corr;
            mrow[r] = mnew[r];
        }
        float psum[8];
        #pragma unroll
        for (int r = 0; r < 8; ++r) psum[r] = 0.0f;
        #pragma unroll
        for (int nt = 0; nt < 4; ++nt) {
            #pragma unroll
            for (int r = 0; r < 8; ++r) {
                float p = exp2f(sc[nt][r] - mrow[r]);
                psum[r] += p;
                sP[w * 16 + half * 8 + r][nt * 16 + l16] = (bf16_t)p;
            }
        }
        #pragma unroll
        for (int r = 0; r < 8; ++r) {
            float ps = psum[r];
            ps += __shfl_xor(ps, 1);
            ps += __shfl_xor(ps, 2);
            ps += __shfl_xor(ps, 4);
            ps += __shfl_xor(ps, 8);
            lrow[r] += ps;
        }

        // ---- ctx += P * V  (sP rows are wave-private; same-wave LDS is in-order)
        #pragma unroll
        for (int kk = 0; kk < 2; ++kk) {
            FragAB pa;
            int prow = w * 16 + l16;
            pa.q[0] = *(const uint4*)&sP[prow][kk * 32 + half * 8];
            pa.q[1] = *(const uint4*)&sP[prow][kk * 32 + 16 + half * 8];
            #pragma unroll
            for (int nt = 0; nt < 4; ++nt) {
                FragAB vf;
                const bf16_t* vrow = (const bf16_t*)&sVt[buf][nt * 16 + l16][0];
                vf.q[0] = *(const uint4*)(vrow + kk * 32 + half * 8);
                vf.q[1] = *(const uint4*)(vrow + kk * 32 + 16 + half * 8);
                o[nt] = wmma_bf16(pa, vf, o[nt]);
            }
        }
    }

    // normalize, write ctx as [b][s][h*64+d] (bf16) for the output GEMM
    const int h = bh & 15, b = bh >> 4;
    #pragma unroll
    for (int r = 0; r < 8; ++r) {
        float inv = 1.0f / lrow[r];
        int srow2 = q0 + w * 16 + half * 8 + r;
        #pragma unroll
        for (int nt = 0; nt < 4; ++nt) {
            int col = h * 64 + nt * 16 + l16;
            Ctx[((size_t)(b * 2048 + srow2)) * 1024 + col] = (bf16_t)(o[nt][r] * inv);
        }
    }
}

// ---------------------------------------------------------------------------
// Output projection: ctx(bf16)[4096,1024] @ Wo^T + bo -> out f32 [4096,1024]
// ---------------------------------------------------------------------------
__global__ __launch_bounds__(256) void dit_gemm_out(
    const bf16_t* __restrict__ X, const bf16_t* __restrict__ W,
    const float* __restrict__ bias, float* __restrict__ Out)
{
    __shared__ __align__(16) bf16_t sX[2][128][32];
    __shared__ __align__(16) bf16_t sW[2][128][32];

    const int m0 = blockIdx.x * 128;
    const int n0 = blockIdx.y * 128;
    const int t    = threadIdx.x;
    const int lane = t & 31;
    const int w    = t >> 5;
    const int wm   = w >> 2;
    const int wn   = w & 3;
    const int l16  = lane & 15;
    const int half = lane >> 4;

    const int srow = t >> 1;
    const int sseg = (t & 1) * 16;
    const bf16_t* gX = X + (size_t)(m0 + srow) * 1024 + sseg;
    const bf16_t* gW = W + (size_t)(n0 + srow) * 1024 + sseg;

    v8f acc[4][2];
    #pragma unroll
    for (int i = 0; i < 4; ++i)
        #pragma unroll
        for (int j = 0; j < 2; ++j)
            #pragma unroll
            for (int e = 0; e < 8; ++e) acc[i][j][e] = 0.0f;

    uint4 rx0 = *(const uint4*)(gX);
    uint4 rx1 = *(const uint4*)(gX + 8);
    uint4 rw0 = *(const uint4*)(gW);
    uint4 rw1 = *(const uint4*)(gW + 8);

    for (int ks = 0; ks < 32; ++ks) {
        const int buf = ks & 1;
        *(uint4*)&sX[buf][srow][sseg]     = rx0;
        *(uint4*)&sX[buf][srow][sseg + 8] = rx1;
        *(uint4*)&sW[buf][srow][sseg]     = rw0;
        *(uint4*)&sW[buf][srow][sseg + 8] = rw1;
        if (ks + 1 < 32) {
            const int k1 = (ks + 1) * 32;
            rx0 = *(const uint4*)(gX + k1);
            rx1 = *(const uint4*)(gX + k1 + 8);
            rw0 = *(const uint4*)(gW + k1);
            rw1 = *(const uint4*)(gW + k1 + 8);
            if (ks + 2 < 32) {
                __builtin_prefetch(gX + k1 + 32, 0, 0);
                __builtin_prefetch(gW + k1 + 32, 0, 0);
            }
        }
        __syncthreads();

        FragAB a[4], b[2];
        #pragma unroll
        for (int i = 0; i < 4; ++i) {
            int row = wm * 64 + i * 16 + l16;
            a[i].q[0] = *(const uint4*)&sX[buf][row][half * 8];
            a[i].q[1] = *(const uint4*)&sX[buf][row][16 + half * 8];
        }
        #pragma unroll
        for (int j = 0; j < 2; ++j) {
            int col = wn * 32 + j * 16 + l16;
            b[j].q[0] = *(const uint4*)&sW[buf][col][half * 8];
            b[j].q[1] = *(const uint4*)&sW[buf][col][16 + half * 8];
        }
        #pragma unroll
        for (int i = 0; i < 4; ++i)
            #pragma unroll
            for (int j = 0; j < 2; ++j)
                acc[i][j] = wmma_bf16(a[i], b[j], acc[i][j]);
    }

    #pragma unroll
    for (int i = 0; i < 4; ++i) {
        #pragma unroll
        for (int j = 0; j < 2; ++j) {
            int col = n0 + wn * 32 + j * 16 + l16;
            float bb = bias[col];
            #pragma unroll
            for (int r = 0; r < 8; ++r) {
                int m = m0 + wm * 64 + i * 16 + half * 8 + r;
                Out[(size_t)m * 1024 + col] = acc[i][j][r] + bb;
            }
        }
    }
}

// ---------------------------------------------------------------------------
// Host launcher
// ---------------------------------------------------------------------------
extern "C" void kernel_launch(void* const* d_in, const int* in_sizes, int n_in,
                              void* d_out, int out_size, void* d_ws, size_t ws_size,
                              hipStream_t stream) {
    const float* hs = (const float*)d_in[0];
    const float* Wq = (const float*)d_in[1];
    const float* bq = (const float*)d_in[2];
    const float* Wk = (const float*)d_in[3];
    const float* bk = (const float*)d_in[4];
    const float* Wv = (const float*)d_in[5];
    const float* bv = (const float*)d_in[6];
    const float* Wo = (const float*)d_in[7];
    const float* bo = (const float*)d_in[8];
    float* out = (float*)d_out;

    char* ws = (char*)d_ws;
    const size_t MB = 1024 * 1024;
    bf16_t* Xbf = (bf16_t*)(ws + 0 * MB);    // 8 MB
    bf16_t* Wqb = (bf16_t*)(ws + 8 * MB);    // 2 MB
    bf16_t* Wkb = (bf16_t*)(ws + 10 * MB);   // 2 MB
    bf16_t* Wvb = (bf16_t*)(ws + 12 * MB);   // 2 MB
    bf16_t* Wob = (bf16_t*)(ws + 14 * MB);   // 2 MB
    bf16_t* Qb  = (bf16_t*)(ws + 16 * MB);   // 8 MB
    bf16_t* Kb  = (bf16_t*)(ws + 24 * MB);   // 8 MB
    bf16_t* Vb  = (bf16_t*)(ws + 32 * MB);   // 8 MB
    bf16_t* Cb  = (bf16_t*)(ws + 40 * MB);   // 8 MB

    const int NX = 4096 * 1024, NW = 1024 * 1024;
    dit_cvt_bf16<<<(NX / 4 + 255) / 256, 256, 0, stream>>>(hs, Xbf, NX / 4);
    dit_cvt_bf16<<<(NW / 4 + 255) / 256, 256, 0, stream>>>(Wq, Wqb, NW / 4);
    dit_cvt_bf16<<<(NW / 4 + 255) / 256, 256, 0, stream>>>(Wk, Wkb, NW / 4);
    dit_cvt_bf16<<<(NW / 4 + 255) / 256, 256, 0, stream>>>(Wv, Wvb, NW / 4);
    dit_cvt_bf16<<<(NW / 4 + 255) / 256, 256, 0, stream>>>(Wo, Wob, NW / 4);

    dit_gemm_qkv<<<dim3(32, 24), 256, 0, stream>>>(Xbf, Wqb, Wkb, Wvb, bq, bk, bv, Qb, Kb, Vb);
    dit_attn<<<dim3(32, 32), 128, 0, stream>>>(Qb, Kb, Vb, Cb);
    dit_gemm_out<<<dim3(32, 8), 256, 0, stream>>>(Cb, Wob, bo, out);
}
